// EDformerPlus_40303973105903
// MI455X (gfx1250) — matrix operand order repaired and checked by reading, not compiled, not run on hardware
//
#include <hip/hip_runtime.h>
#include <hip/hip_bf16.h>
#include <math.h>

// ---------------- model constants ----------------
#define B_    2
#define N_    8192
#define M_    (B_ * N_)          // 16384 rows
#define KB_   9                  // K_BALL
#define KL_   16                 // K_L
#define D_    32
#define DI_   64
#define DS_   64
#define DR_   2
#define DC_   4
#define EPS_  1e-5f
#define NCH_  64                 // scan chunks
#define CHUNK_ 128               // N_/NCH_

static constexpr float RADIUS2_  = (float)((5.0 / 260.0) * (5.0 / 260.0));
static constexpr float INV_ATTN_ = 0.17677669529663687f;  // 1/sqrt(32)

typedef __attribute__((ext_vector_type(16))) _Float16 v16h;
typedef __attribute__((ext_vector_type(8)))  float    v8f;

// ---------------- small device helpers ----------------
__device__ __forceinline__ float wsum32(float v) {
  v += __shfl_xor(v, 1);
  v += __shfl_xor(v, 2);
  v += __shfl_xor(v, 4);
  v += __shfl_xor(v, 8);
  v += __shfl_xor(v, 16);
  return v;
}
__device__ __forceinline__ float gelu_f(float x) {
  float x3 = x * x * x;
  return 0.5f * x * (1.f + tanhf(0.7978845608028654f * (x + 0.044715f * x3)));
}
__device__ __forceinline__ float silu_f(float x) {
  return x / (1.f + __expf(-x));
}

// ---------------- KNN (ball, 2D, K=9) -> e (M,18) ----------------
__global__ void __launch_bounds__(256) knn_ball_kernel(const float* __restrict__ xytp,
                                                       float* __restrict__ e) {
  const int bpb = N_ / 256;
  int b = blockIdx.x / bpb;
  int i = (blockIdx.x % bpb) * 256 + threadIdx.x;
  const float* xb = xytp + (size_t)b * N_ * 4;
  float qx = xb[i * 4 + 1], qy = xb[i * 4 + 2];
  float bd[KB_]; int bi[KB_];
#pragma unroll
  for (int k = 0; k < KB_; ++k) { bd[k] = 3.4e38f; bi[k] = i; }
  __shared__ float sx[256], sy[256];
  for (int j0 = 0; j0 < N_; j0 += 256) {
    __syncthreads();
    int j = j0 + threadIdx.x;
    sx[threadIdx.x] = xb[j * 4 + 1];
    sy[threadIdx.x] = xb[j * 4 + 2];
    __syncthreads();
    if (j0 + 256 < N_) {
      // prefetch next candidate tile while scanning this one (global_prefetch_b8)
      __builtin_prefetch(&xb[(size_t)(j0 + 256 + threadIdx.x) * 4], 0, 0);
    }
    for (int jj = 0; jj < 256; ++jj) {
      float dx = qx - sx[jj], dy = qy - sy[jj];
      float d2 = dx * dx + dy * dy;
      if (d2 < bd[KB_ - 1]) {
        float cd = d2; int ci = j0 + jj;
#pragma unroll
        for (int p = 0; p < KB_; ++p) {
          if (cd < bd[p]) { float td = bd[p]; int ti = bi[p]; bd[p] = cd; bi[p] = ci; cd = td; ci = ti; }
        }
      }
    }
  }
  float* er = e + ((size_t)b * N_ + i) * (2 * KB_);
#pragma unroll
  for (int k = 0; k < KB_; ++k) {
    int id = (bd[k] <= RADIUS2_) ? bi[k] : i;
    er[2 * k + 0] = qx - xb[id * 4 + 1];
    er[2 * k + 1] = qy - xb[id * 4 + 2];
  }
}

// ---------------- KNN (3D xyt, K=16) -> idxl (M,16) ----------------
__global__ void __launch_bounds__(256) knn_l_kernel(const float* __restrict__ xytp,
                                                    int* __restrict__ idxl) {
  const int bpb = N_ / 256;
  int b = blockIdx.x / bpb;
  int i = (blockIdx.x % bpb) * 256 + threadIdx.x;
  const float* xb = xytp + (size_t)b * N_ * 4;
  float qt = xb[i * 4 + 0], qx = xb[i * 4 + 1], qy = xb[i * 4 + 2];
  float bd[KL_]; int bi[KL_];
#pragma unroll
  for (int k = 0; k < KL_; ++k) { bd[k] = 3.4e38f; bi[k] = i; }
  __shared__ float st[256], sx[256], sy[256];
  for (int j0 = 0; j0 < N_; j0 += 256) {
    __syncthreads();
    int j = j0 + threadIdx.x;
    st[threadIdx.x] = xb[j * 4 + 0];
    sx[threadIdx.x] = xb[j * 4 + 1];
    sy[threadIdx.x] = xb[j * 4 + 2];
    __syncthreads();
    if (j0 + 256 < N_) {
      __builtin_prefetch(&xb[(size_t)(j0 + 256 + threadIdx.x) * 4], 0, 0);
    }
    for (int jj = 0; jj < 256; ++jj) {
      float dt0 = qt - st[jj], dx = qx - sx[jj], dy = qy - sy[jj];
      float d2 = dt0 * dt0 + dx * dx + dy * dy;
      if (d2 < bd[KL_ - 1]) {
        float cd = d2; int ci = j0 + jj;
#pragma unroll
        for (int p = 0; p < KL_; ++p) {
          if (cd < bd[p]) { float td = bd[p]; int ti = bi[p]; bd[p] = cd; bi[p] = ci; cd = td; ci = ti; }
        }
      }
    }
  }
  int* ir = idxl + ((size_t)b * N_ + i) * KL_;
#pragma unroll
  for (int k = 0; k < KL_; ++k) ir[k] = bi[k];
}

// ---------------- generic WMMA GEMM: Y(M x Nout) = act(A(M x K) @ W(K x Nout) + bias) --------
// Templated on NKC = Kpad/32 so all fragment loops are static:
//   * A fragments (per K-chunk) gathered ONCE per wave, reused for all column tiles
//   * all B-chunk ds_loads for a tile issued before the WMMA chain
// 256 threads = 8 waves; each wave owns a 16-row stripe; block covers 128 rows.
template <int NKC>
__global__ void __launch_bounds__(256) gemm_wmma_kernel(const float* __restrict__ A,
                                                        const float* __restrict__ W,
                                                        const float* __restrict__ bias,
                                                        float* __restrict__ Y,
                                                        int K, int Nout, int Npad,
                                                        int act) {
  constexpr int Kpad = NKC * 32;
  extern __shared__ _Float16 sm[];
  _Float16* As = sm;                     // [128][Kpad]
  _Float16* Ws = sm + 128 * Kpad;        // [Npad][Kpad]  (Ws[n*Kpad+k] = W[k][n])
  int tid  = threadIdx.x;
  int row0 = blockIdx.x * 128;

  for (int p = tid; p < Npad * Kpad; p += 256) {
    int n = p / Kpad, k = p - n * Kpad;
    float v = (n < Nout && k < K) ? W[(size_t)k * Nout + n] : 0.f;
    Ws[p] = (_Float16)v;
  }
  for (int p = tid; p < 128 * Kpad; p += 256) {
    int m = p / Kpad, k = p - m * Kpad;
    float v = (k < K) ? A[(size_t)(row0 + m) * K + k] : 0.f;
    As[p] = (_Float16)v;
  }
  __syncthreads();

  int w = tid >> 5, lane = tid & 31;
  int l15 = lane & 15;
  const _Float16* Arow = As + ((size_t)((w << 4) + l15)) * Kpad;

  union U { v16h v; unsigned u[8]; };
  // gather A fragments once (layout per ISA: lo lanes hold K 0-7/16-23, hi lanes 8-15/24-31)
  U ua[NKC];
#pragma unroll
  for (int kc = 0; kc < NKC; ++kc) {
    int ka = (kc << 5) + ((lane & 16) ? 8 : 0);
#pragma unroll
    for (int q = 0; q < 8; ++q) {
      int koff = (q < 4) ? (2 * q) : (16 + 2 * (q - 4));
      ua[kc].u[q] = *(const unsigned*)(Arow + ka + koff);
    }
  }

  int ntiles = Npad >> 4;
  for (int nt = 0; nt < ntiles; ++nt) {
    int ncol = (nt << 4) + l15;
    const _Float16* Wrow = Ws + (size_t)ncol * Kpad;
    float bv = 0.f;
    if (bias != nullptr && ncol < Nout) bv = bias[ncol];
    v8f acc;
#pragma unroll
    for (int r = 0; r < 8; ++r) acc[r] = bv;

    // issue all B-chunk loads (B f16 layout: lo lanes K 0-7/16-23, hi lanes 8-15/24-31)
    U ub[NKC];
#pragma unroll
    for (int kc = 0; kc < NKC; ++kc) {
      int kb = (kc << 5) + ((lane & 16) ? 8 : 0);
#pragma unroll
      for (int q = 0; q < 8; ++q) {
        ub[kc].u[q] = *(const unsigned*)(Wrow + kb + ((q < 4) ? (2 * q) : (16 + 2 * (q - 4))));
      }
    }
    // WMMA chain
#pragma unroll
    for (int kc = 0; kc < NKC; ++kc) {
      acc = __builtin_amdgcn_wmma_f32_16x16x32_f16(false, ua[kc].v, false, ub[kc].v,
                                                   (short)0, acc, false, false);
    }

    if (ncol < Nout) {
#pragma unroll
      for (int r = 0; r < 8; ++r) {
        int m = (w << 4) + r + ((lane & 16) ? 8 : 0);
        float v = acc[r];
        if (act == 1) v = gelu_f(v);
        Y[(size_t)(row0 + m) * Nout + ncol] = v;
      }
    }
  }
}

// ---------------- cat = [Fsp | Fte] ----------------
__global__ void __launch_bounds__(256) cat_kernel(const float* __restrict__ Fsp,
                                                  const float* __restrict__ xytp,
                                                  const float* __restrict__ te_pw,
                                                  const float* __restrict__ te_pb,
                                                  const float* __restrict__ te_nw,
                                                  const float* __restrict__ te_nb,
                                                  float* __restrict__ cat) {
  int idx = blockIdx.x * 256 + threadIdx.x;     // M_*32
  int r = idx >> 5, c = idx & 31;
  float ts  = xytp[(size_t)r * 4 + 0];
  float pol = xytp[(size_t)r * 4 + 3];
  float fte = (ts * te_pw[c] + te_pb[c]) * pol + (ts * te_nw[c] + te_nb[c]) * (1.f - pol);
  cat[(size_t)r * 64 + c]      = Fsp[idx];
  cat[(size_t)r * 64 + 32 + c] = fte;
}

// ---------------- LayerNorm over last dim 32 (one wave per row) ----------------
__global__ void __launch_bounds__(256) ln_kernel(const float* __restrict__ X,
                                                 const float* __restrict__ G,
                                                 const float* __restrict__ Bt,
                                                 float* __restrict__ Y) {
  int w = threadIdx.x >> 5, lane = threadIdx.x & 31;
  int row = blockIdx.x * 8 + w;
  float x = X[(size_t)row * 32 + lane];
  float m = wsum32(x) * (1.f / 32.f);
  float d = x - m;
  float v = wsum32(d * d) * (1.f / 32.f);
  Y[(size_t)row * 32 + lane] = d * rsqrtf(v + EPS_) * G[lane] + Bt[lane];
}

// ---------------- elementwise add ----------------
__global__ void __launch_bounds__(256) add_kernel(const float* __restrict__ a,
                                                  const float* __restrict__ b,
                                                  float* __restrict__ c, int n) {
  int i = blockIdx.x * 256 + threadIdx.x;
  if (i < n) c[i] = a[i] + b[i];
}

// ---------------- BN stats for delta_pe (sums of h1 and h1^2 per 4 channels) ---------
__global__ void zero8_kernel(float* s) { if (threadIdx.x < 8) s[threadIdx.x] = 0.f; }

__global__ void __launch_bounds__(256) bnstat_kernel(const float* __restrict__ xytp,
                                                     const int* __restrict__ idxl,
                                                     const float* __restrict__ w1,
                                                     const float* __restrict__ b1,
                                                     float* __restrict__ stat) {
  __shared__ float ss[8];
  if (threadIdx.x < 8) ss[threadIdx.x] = 0.f;
  __syncthreads();
  int idx = blockIdx.x * 256 + threadIdx.x;   // B*N*16
  int r = idx >> 4, k = idx & 15;
  int b = r >> 13, i = r & (N_ - 1);
  int j = idxl[(size_t)r * KL_ + k];
  const float* xb = xytp + (size_t)b * N_ * 4;
  float dp[4];
#pragma unroll
  for (int c = 0; c < 4; ++c) dp[c] = xb[(size_t)i * 4 + c] - xb[(size_t)j * 4 + c];
  float h[4], h2[4];
#pragma unroll
  for (int q = 0; q < 4; ++q) {
    float v = b1[q];
#pragma unroll
    for (int jj = 0; jj < 4; ++jj) v += dp[jj] * w1[jj * 4 + q];
    h[q] = v; h2[q] = v * v;
  }
  int lane = threadIdx.x & 31;
#pragma unroll
  for (int q = 0; q < 4; ++q) { h[q] = wsum32(h[q]); h2[q] = wsum32(h2[q]); }
  if (lane == 0) {
#pragma unroll
    for (int q = 0; q < 4; ++q) { atomicAdd(&ss[q], h[q]); atomicAdd(&ss[4 + q], h2[q]); }
  }
  __syncthreads();
  if (threadIdx.x < 8) atomicAdd(&stat[threadIdx.x], ss[threadIdx.x]);
}

// ---------------- local attention (one wave = one point, lane = channel) ----------------
__global__ void __launch_bounds__(256) attn_kernel(const float* __restrict__ lt,
                                                   const int* __restrict__ idxl,
                                                   const float* __restrict__ xytp,
                                                   const float* __restrict__ pe_w1,
                                                   const float* __restrict__ pe_b1,
                                                   const float* __restrict__ pe_bn_g,
                                                   const float* __restrict__ pe_bn_b,
                                                   const float* __restrict__ pe_w2,
                                                   const float* __restrict__ pe_b2,
                                                   const float* __restrict__ lx_g,
                                                   const float* __restrict__ lx_b,
                                                   const float* __restrict__ stat,
                                                   float* __restrict__ out) {
  int w = threadIdx.x >> 5, lane = threadIdx.x & 31;
  int row = blockIdx.x * 8 + w;
  int b = row >> 13, i = row & (N_ - 1);

  const float inv_cnt = 1.f / (float)(B_ * N_ * KL_);
  float mu[4], rstd[4], w1r[16], b1r[4], bg[4], bb[4], w2c[4];
#pragma unroll
  for (int q = 0; q < 4; ++q) {
    float m = stat[q] * inv_cnt;
    float v = stat[4 + q] * inv_cnt - m * m;
    mu[q] = m; rstd[q] = rsqrtf(v + EPS_);
    b1r[q] = pe_b1[q]; bg[q] = pe_bn_g[q]; bb[q] = pe_bn_b[q];
    w2c[q] = pe_w2[q * 32 + lane];
  }
#pragma unroll
  for (int q = 0; q < 16; ++q) w1r[q] = pe_w1[q];
  float b2c = pe_b2[lane];
  float lg = lx_g[lane], lb = lx_b[lane];

  const float* xb  = xytp + (size_t)b * N_ * 4;
  const float* ltb = lt + (size_t)b * N_ * 96;
  float pi0 = xb[(size_t)i * 4 + 0], pi1 = xb[(size_t)i * 4 + 1];
  float pi2 = xb[(size_t)i * 4 + 2], pi3 = xb[(size_t)i * 4 + 3];
  float varphi = ltb[(size_t)i * 96 + lane];
  const int* ir = idxl + (size_t)row * KL_;

  float pre[KL_], val[KL_];
#pragma unroll
  for (int k = 0; k < KL_; ++k) {
    int j = ir[k];
    float d0 = pi0 - xb[(size_t)j * 4 + 0];
    float d1 = pi1 - xb[(size_t)j * 4 + 1];
    float d2 = pi2 - xb[(size_t)j * 4 + 2];
    float d3 = pi3 - xb[(size_t)j * 4 + 3];
    float dpe = b2c;
#pragma unroll
    for (int q = 0; q < 4; ++q) {
      float h1 = b1r[q] + d0 * w1r[0 * 4 + q] + d1 * w1r[1 * 4 + q]
                        + d2 * w1r[2 * 4 + q] + d3 * w1r[3 * 4 + q];
      h1 = (h1 - mu[q]) * rstd[q] * bg[q] + bb[q];
      h1 = fmaxf(h1, 0.f);
      dpe += h1 * w2c[q];
    }
    float pr = varphi - ltb[(size_t)j * 96 + 32 + lane] + dpe;
    float mean = wsum32(pr) * (1.f / 32.f);
    float dd = pr - mean;
    float var = wsum32(dd * dd) * (1.f / 32.f);
    pre[k] = (dd * rsqrtf(var + EPS_) * lg + lb) * INV_ATTN_;
    val[k] = ltb[(size_t)j * 96 + 64 + lane] + dpe;
  }
  float mx = -3.4e38f;
#pragma unroll
  for (int k = 0; k < KL_; ++k) mx = fmaxf(mx, pre[k]);
  float se = 0.f, acc = 0.f;
#pragma unroll
  for (int k = 0; k < KL_; ++k) {
    float ev = __expf(pre[k] - mx);
    se += ev; acc += ev * val[k];
  }
  out[(size_t)row * 32 + lane] = acc / se;
}

// ---------------- mamba: causal depthwise conv (4 taps) + SiLU ----------------
__global__ void __launch_bounds__(256) conv_kernel(const float* __restrict__ xz,
                                                   const float* __restrict__ cw,
                                                   const float* __restrict__ cb,
                                                   float* __restrict__ u) {
  int idx = blockIdx.x * 256 + threadIdx.x;   // M_*64
  int r = idx >> 6, d = idx & 63;
  int b = r >> 13, n = r & (N_ - 1);
  float acc = cb[d];
#pragma unroll
  for (int k = 0; k < DC_; ++k) {
    int nn = n + k - (DC_ - 1);
    if (nn >= 0) acc += xz[((size_t)(b * N_ + nn)) * 128 + d] * cw[d * DC_ + k];
  }
  u[idx] = silu_f(acc);
}

// ---------------- dt = softplus(xdbl[:,:2] @ m_dt_w + m_dt_b) ----------------
__global__ void __launch_bounds__(256) dt_kernel(const float* __restrict__ xdbl,
                                                 const float* __restrict__ dw,
                                                 const float* __restrict__ db,
                                                 float* __restrict__ dt) {
  int idx = blockIdx.x * 256 + threadIdx.x;   // M_*64
  int r = idx >> 6, d = idx & 63;
  const float* xr = xdbl + (size_t)r * 130;
  float x = xr[0] * dw[d] + xr[1] * dw[64 + d] + db[d];
  dt[idx] = (x > 20.f) ? x : log1pf(__expf(x));
}

// ---------------- chunked selective scan ----------------
// thread t: d = t>>4, s in [(t&15)*4, +4). h_t = h*exp(dt*A) + dt*B*u.
__global__ void __launch_bounds__(1024) scanA_kernel(const float* __restrict__ dt,
                                                     const float* __restrict__ xdbl,
                                                     const float* __restrict__ u,
                                                     const float* __restrict__ Alog,
                                                     float* __restrict__ hend,
                                                     float* __restrict__ aprod) {
  int b = blockIdx.x / NCH_, ch = blockIdx.x % NCH_;
  int t = threadIdx.x;
  int d = t >> 4, s0 = (t & 15) << 2;
  float Ar[4];
#pragma unroll
  for (int q = 0; q < 4; ++q) Ar[q] = -__expf(Alog[d * 64 + s0 + q]);
  float h[4] = {0, 0, 0, 0}, ap[4] = {1, 1, 1, 1};
  size_t rb = (size_t)b * N_ + (size_t)ch * CHUNK_;
  for (int n = 0; n < CHUNK_; ++n) {
    size_t r = rb + n;
    float dtv = dt[r * 64 + d];
    float uv  = u[r * 64 + d];
    const float* xr = xdbl + r * 130;
    float du = dtv * uv;
#pragma unroll
    for (int q = 0; q < 4; ++q) {
      float dA = __expf(dtv * Ar[q]);
      h[q] = h[q] * dA + du * xr[2 + s0 + q];
      ap[q] *= dA;
    }
  }
  size_t o = (size_t)blockIdx.x * 4096 + d * 64 + s0;
#pragma unroll
  for (int q = 0; q < 4; ++q) { hend[o + q] = h[q]; aprod[o + q] = ap[q]; }
}

__global__ void __launch_bounds__(256) scanB_kernel(const float* __restrict__ hend,
                                                    const float* __restrict__ aprod,
                                                    float* __restrict__ hinit) {
  int idx = blockIdx.x * 256 + threadIdx.x;    // B_*4096
  int b = idx >> 12, es = idx & 4095;
  float hi = 0.f;
  for (int ch = 0; ch < NCH_; ++ch) {
    size_t o = (size_t)(b * NCH_ + ch) * 4096 + es;
    hinit[o] = hi;
    hi = hi * aprod[o] + hend[o];
  }
}

__global__ void __launch_bounds__(1024) scanC_kernel(const float* __restrict__ dt,
                                                     const float* __restrict__ xdbl,
                                                     const float* __restrict__ u,
                                                     const float* __restrict__ Alog,
                                                     const float* __restrict__ hinit,
                                                     float* __restrict__ ys) {
  int b = blockIdx.x / NCH_, ch = blockIdx.x % NCH_;
  int t = threadIdx.x;
  int d = t >> 4, s0 = (t & 15) << 2;
  float Ar[4];
#pragma unroll
  for (int q = 0; q < 4; ++q) Ar[q] = -__expf(Alog[d * 64 + s0 + q]);
  size_t o = (size_t)blockIdx.x * 4096 + d * 64 + s0;
  float h[4];
#pragma unroll
  for (int q = 0; q < 4; ++q) h[q] = hinit[o + q];
  size_t rb = (size_t)b * N_ + (size_t)ch * CHUNK_;
  for (int n = 0; n < CHUNK_; ++n) {
    size_t r = rb + n;
    float dtv = dt[r * 64 + d];
    float uv  = u[r * 64 + d];
    const float* xr = xdbl + r * 130;
    float du = dtv * uv;
    float p = 0.f;
#pragma unroll
    for (int q = 0; q < 4; ++q) {
      float dA = __expf(dtv * Ar[q]);
      h[q] = h[q] * dA + du * xr[2 + s0 + q];
      p += h[q] * xr[66 + s0 + q];
    }
    p += __shfl_xor(p, 1);
    p += __shfl_xor(p, 2);
    p += __shfl_xor(p, 4);
    p += __shfl_xor(p, 8);
    if ((t & 15) == 0) ys[r * 64 + d] = p;
  }
}

// ---------------- y = (ys + u*D) * silu(z) ----------------
__global__ void __launch_bounds__(256) ymod_kernel(const float* __restrict__ ys,
                                                   const float* __restrict__ u,
                                                   const float* __restrict__ xz,
                                                   const float* __restrict__ mD,
                                                   float* __restrict__ ym) {
  int idx = blockIdx.x * 256 + threadIdx.x;   // M_*64
  int r = idx >> 6, d = idx & 63;
  float z = xz[(size_t)r * 128 + 64 + d];
  ym[idx] = (ys[idx] + u[idx] * mD[d]) * silu_f(z);
}

// ---------------- head: out = F3 @ head_w + head_b ----------------
__global__ void __launch_bounds__(256) head_kernel(const float* __restrict__ F3,
                                                   const float* __restrict__ hw,
                                                   const float* __restrict__ hb,
                                                   float* __restrict__ out) {
  int w = threadIdx.x >> 5, lane = threadIdx.x & 31;
  int row = blockIdx.x * 8 + w;
  float v = F3[(size_t)row * 32 + lane] * hw[lane];
  v = wsum32(v);
  if (lane == 0) out[row] = v + hb[0];
}

// ---------------- host side ----------------
static inline void gemm_launch(const float* A, const float* W, const float* bias, float* Y,
                               int K, int Nout, int act, hipStream_t s) {
  int Kpad = (K + 31) & ~31;
  int Npad = (Nout + 15) & ~15;
  size_t sh = (size_t)(128 * Kpad + Npad * Kpad) * 2;  // f16 elements
  dim3 g(M_ / 128), blk(256);
  switch (Kpad >> 5) {
    case 1: gemm_wmma_kernel<1><<<g, blk, sh, s>>>(A, W, bias, Y, K, Nout, Npad, act); break;
    case 2: gemm_wmma_kernel<2><<<g, blk, sh, s>>>(A, W, bias, Y, K, Nout, Npad, act); break;
    default: gemm_wmma_kernel<4><<<g, blk, sh, s>>>(A, W, bias, Y, K, Nout, Npad, act); break;
  }
}

extern "C" void kernel_launch(void* const* d_in, const int* in_sizes, int n_in,
                              void* d_out, int out_size, void* d_ws, size_t ws_size,
                              hipStream_t stream) {
  const float* xytp    = (const float*)d_in[0];
  const float* se_w1   = (const float*)d_in[1];
  const float* se_b1   = (const float*)d_in[2];
  const float* se_w2   = (const float*)d_in[3];
  const float* se_b2   = (const float*)d_in[4];
  const float* te_pw   = (const float*)d_in[5];
  const float* te_pb   = (const float*)d_in[6];
  const float* te_nw   = (const float*)d_in[7];
  const float* te_nb   = (const float*)d_in[8];
  const float* proj_w  = (const float*)d_in[9];
  const float* emb_g   = (const float*)d_in[10];
  const float* emb_b   = (const float*)d_in[11];
  const float* n1_g    = (const float*)d_in[12];
  const float* n1_b    = (const float*)d_in[13];
  const float* n2_g    = (const float*)d_in[14];
  const float* n2_b    = (const float*)d_in[15];
  const float* pe_w1   = (const float*)d_in[16];
  const float* pe_b1   = (const float*)d_in[17];
  const float* pe_bn_g = (const float*)d_in[18];
  const float* pe_bn_b = (const float*)d_in[19];
  const float* pe_w2   = (const float*)d_in[20];
  const float* pe_b2   = (const float*)d_in[21];
  const float* lt_w    = (const float*)d_in[22];
  const float* lt_b    = (const float*)d_in[23];
  const float* lx_g    = (const float*)d_in[24];
  const float* lx_b    = (const float*)d_in[25];
  const float* m_in_w  = (const float*)d_in[26];
  const float* m_cw    = (const float*)d_in[27];
  const float* m_cb    = (const float*)d_in[28];
  const float* m_xw    = (const float*)d_in[29];
  const float* m_dtw   = (const float*)d_in[30];
  const float* m_dtb   = (const float*)d_in[31];
  const float* m_Alog  = (const float*)d_in[32];
  const float* m_D     = (const float*)d_in[33];
  const float* m_ow    = (const float*)d_in[34];
  const float* mlp_w1  = (const float*)d_in[35];
  const float* mlp_b1  = (const float*)d_in[36];
  const float* mlp_w2  = (const float*)d_in[37];
  const float* mlp_b2  = (const float*)d_in[38];
  const float* head_w  = (const float*)d_in[39];
  const float* head_b  = (const float*)d_in[40];

  // workspace bump allocator
  char* wp = (char*)d_ws;
  auto alloc = [&](size_t nbytes) -> void* {
    char* p = wp;
    wp += (nbytes + 255) & ~(size_t)255;
    return (void*)p;
  };
  float* e     = (float*)alloc((size_t)M_ * 18 * 4);
  int*   idxl  = (int*)  alloc((size_t)M_ * 16 * 4);
  float* act1  = (float*)alloc((size_t)M_ * 64 * 4);
  float* Fsp   = (float*)alloc((size_t)M_ * 32 * 4);
  float* cat   = (float*)alloc((size_t)M_ * 64 * 4);
  float* P     = (float*)alloc((size_t)M_ * 32 * 4);
  float* F     = (float*)alloc((size_t)M_ * 32 * 4);
  float* fn    = (float*)alloc((size_t)M_ * 32 * 4);
  float* lt    = (float*)alloc((size_t)M_ * 96 * 4);
  float* stat  = (float*)alloc(8 * 4);
  float* attn  = (float*)alloc((size_t)M_ * 32 * 4);
  float* xz    = (float*)alloc((size_t)M_ * 128 * 4);
  float* u     = (float*)alloc((size_t)M_ * 64 * 4);
  float* xdbl  = (float*)alloc((size_t)M_ * 130 * 4);
  float* dtv   = (float*)alloc((size_t)M_ * 64 * 4);
  float* hend  = (float*)alloc((size_t)B_ * NCH_ * 4096 * 4);
  float* aprod = (float*)alloc((size_t)B_ * NCH_ * 4096 * 4);
  float* hinit = (float*)alloc((size_t)B_ * NCH_ * 4096 * 4);
  float* ys    = (float*)alloc((size_t)M_ * 64 * 4);
  float* ymod  = (float*)alloc((size_t)M_ * 64 * 4);
  float* mO    = (float*)alloc((size_t)M_ * 32 * 4);
  float* F2    = (float*)alloc((size_t)M_ * 32 * 4);
  float* h2    = (float*)alloc((size_t)M_ * 32 * 4);
  float* mlp1  = (float*)alloc((size_t)M_ * 128 * 4);
  float* mlpo  = (float*)alloc((size_t)M_ * 32 * 4);
  float* F3    = (float*)alloc((size_t)M_ * 32 * 4);

  dim3 b256(256);
  // 1) KNN passes
  knn_ball_kernel<<<dim3(B_ * (N_ / 256)), b256, 0, stream>>>(xytp, e);
  knn_l_kernel<<<dim3(B_ * (N_ / 256)), b256, 0, stream>>>(xytp, idxl);
  // 2) spatial embedding MLP (WMMA)
  gemm_launch(e, se_w1, se_b1, act1, 18, 64, /*gelu*/1, stream);
  gemm_launch(act1, se_w2, se_b2, Fsp, 64, 32, 0, stream);
  // 3) temporal embedding + concat, proj, emb LN
  cat_kernel<<<dim3(M_ * 32 / 256), b256, 0, stream>>>(Fsp, xytp, te_pw, te_pb, te_nw, te_nb, cat);
  gemm_launch(cat, proj_w, nullptr, P, 64, 32, 0, stream);
  ln_kernel<<<dim3(M_ / 8), b256, 0, stream>>>(P, emb_g, emb_b, F);
  // 4) norm1 + lt projection
  ln_kernel<<<dim3(M_ / 8), b256, 0, stream>>>(F, n1_g, n1_b, fn);
  gemm_launch(fn, lt_w, lt_b, lt, 32, 96, 0, stream);
  // 5) positional-encoding batchnorm stats + local attention
  zero8_kernel<<<dim3(1), dim3(8), 0, stream>>>(stat);
  bnstat_kernel<<<dim3(B_ * N_ * KL_ / 256), b256, 0, stream>>>(xytp, idxl, pe_w1, pe_b1, stat);
  attn_kernel<<<dim3(M_ / 8), b256, 0, stream>>>(lt, idxl, xytp, pe_w1, pe_b1, pe_bn_g, pe_bn_b,
                                                 pe_w2, pe_b2, lx_g, lx_b, stat, attn);
  // 6) mamba
  gemm_launch(attn, m_in_w, nullptr, xz, 32, 128, 0, stream);
  conv_kernel<<<dim3(M_ * 64 / 256), b256, 0, stream>>>(xz, m_cw, m_cb, u);
  gemm_launch(u, m_xw, nullptr, xdbl, 64, 130, 0, stream);
  dt_kernel<<<dim3(M_ * 64 / 256), b256, 0, stream>>>(xdbl, m_dtw, m_dtb, dtv);
  scanA_kernel<<<dim3(B_ * NCH_), dim3(1024), 0, stream>>>(dtv, xdbl, u, m_Alog, hend, aprod);
  scanB_kernel<<<dim3(B_ * 4096 / 256), b256, 0, stream>>>(hend, aprod, hinit);
  scanC_kernel<<<dim3(B_ * NCH_), dim3(1024), 0, stream>>>(dtv, xdbl, u, m_Alog, hinit, ys);
  ymod_kernel<<<dim3(M_ * 64 / 256), b256, 0, stream>>>(ys, u, xz, m_D, ymod);
  gemm_launch(ymod, m_ow, nullptr, mO, 64, 32, 0, stream);
  add_kernel<<<dim3(M_ * 32 / 256), b256, 0, stream>>>(mO, F, F2, M_ * 32);
  // 7) MLP block
  ln_kernel<<<dim3(M_ / 8), b256, 0, stream>>>(F2, n2_g, n2_b, h2);
  gemm_launch(h2, mlp_w1, mlp_b1, mlp1, 32, 128, /*gelu*/1, stream);
  gemm_launch(mlp1, mlp_w2, mlp_b2, mlpo, 128, 32, 0, stream);
  add_kernel<<<dim3(M_ * 32 / 256), b256, 0, stream>>>(mlpo, F2, F3, M_ * 32);
  // 8) head
  head_kernel<<<dim3(M_ / 8), b256, 0, stream>>>(F3, head_w, head_b, (float*)d_out);
}
